// PointerNetwork_with_mask_68049461838609
// MI455X (gfx1250) — matrix operand order repaired and checked by reading, not compiled, not run on hardware
//
#include <hip/hip_runtime.h>
#include <hip/hip_bf16.h>
#include <math.h>

// Problem dims (fixed by the reference)
constexpr int kB = 2048;   // batch
constexpr int kS = 50;     // seq len
constexpr int kE = 256;    // embedding
constexpr int kH = 512;    // hidden
constexpr int kW = 256;    // attention width
constexpr int kL = 50;     // decode steps
constexpr int kG = 4 * kH; // gate width (2048)

typedef __attribute__((ext_vector_type(16))) __bf16 v16bf;
typedef __attribute__((ext_vector_type(8)))  __bf16 v8bf;
typedef __attribute__((ext_vector_type(8)))  float  v8f;

// ---------------------------------------------------------------------------
// WMMA bf16 GEMM: C[M,N] = A1[M,K1] @ B1[N,K1]^T + A2[M,K2] @ B2[N,K2]^T + bias
// (B matrices stored row-major [N,K] like torch weights -> we read 16
//  contiguous K per lane, k = 16*lhalf + s, the same order as the A fragments.)
// Block = 256 threads = 8 wave32; wave grid 4(M) x 2(N); wave tile 32x64;
// block tile 128x128. M,N multiples of 128; K multiples of 32. No LDS: all
// operands are L2-resident on MI455X (192MB L2).
// ---------------------------------------------------------------------------
__device__ inline void wmma_kloop(const __bf16* __restrict__ A, int lda,
                                  const __bf16* __restrict__ Bw, int ldb, int K,
                                  int mwave, int nwave, int lm, int lhalf,
                                  v8f acc[2][4])
{
    const int base8 = lhalf * 16;   // A: lanes 16-31 hold K=16..31
    const int kb16  = lhalf * 16;   // B: lanes 16-31 hold K=16..31
    for (int k0 = 0; k0 < K; k0 += 32) {
        v16bf afrag[2];
#pragma unroll
        for (int mi = 0; mi < 2; ++mi) {
            const __bf16* ap = A + (size_t)(mwave + mi * 16 + lm) * lda + k0 + base8;
            v8bf lo = *(const v8bf*)(ap);        // K = base8..base8+7
            v8bf hi = *(const v8bf*)(ap + 8);    // K = base8+8..base8+15
            v16bf a;
#pragma unroll
            for (int j = 0; j < 8; ++j) { a[j] = lo[j]; a[j + 8] = hi[j]; }
            afrag[mi] = a;
        }
#pragma unroll
        for (int ni = 0; ni < 4; ++ni) {
            const __bf16* bp = Bw + (size_t)(nwave + ni * 16 + lm) * ldb + k0 + kb16;
            v16bf bfrag = *(const v16bf*)bp;     // 16 contiguous K values
#pragma unroll
            for (int mi = 0; mi < 2; ++mi)
                acc[mi][ni] = __builtin_amdgcn_wmma_f32_16x16x32_bf16(
                    false, afrag[mi], false, bfrag, (short)0, acc[mi][ni],
                    false, false);
        }
    }
}

__global__ __launch_bounds__(256)
void gemm_bf16_wmma(const __bf16* __restrict__ A1, int lda1,
                    const __bf16* __restrict__ B1, int ldb1, int K1,
                    const __bf16* __restrict__ A2, int lda2,
                    const __bf16* __restrict__ B2, int ldb2, int K2,
                    const float* __restrict__ bias1,
                    const float* __restrict__ bias2,
                    float* __restrict__ C, int ldc)
{
    const int lane  = threadIdx.x & 31;
    const int wave  = threadIdx.x >> 5;
    const int waveM = wave >> 1;                 // 0..3
    const int waveN = wave & 1;                  // 0..1
    const int mwave = blockIdx.y * 128 + waveM * 32;
    const int nwave = blockIdx.x * 128 + waveN * 64;
    const int lm    = lane & 15;
    const int lhalf = lane >> 4;

    v8f acc[2][4];
#pragma unroll
    for (int mi = 0; mi < 2; ++mi)
#pragma unroll
        for (int ni = 0; ni < 4; ++ni)
#pragma unroll
            for (int r = 0; r < 8; ++r) acc[mi][ni][r] = 0.0f;

    if (K1 > 0) wmma_kloop(A1, lda1, B1, ldb1, K1, mwave, nwave, lm, lhalf, acc);
    if (K2 > 0) wmma_kloop(A2, lda2, B2, ldb2, K2, mwave, nwave, lm, lhalf, acc);

    // C/D layout: VGPR r -> M = r + 8*lhalf; lane%16 -> N
#pragma unroll
    for (int ni = 0; ni < 4; ++ni) {
        const int n = nwave + ni * 16 + lm;
        float bv = 0.0f;
        if (bias1) bv += bias1[n];
        if (bias2) bv += bias2[n];
#pragma unroll
        for (int mi = 0; mi < 2; ++mi) {
            const int mbase = mwave + mi * 16 + lhalf * 8;
#pragma unroll
            for (int r = 0; r < 8; ++r)
                C[(size_t)(mbase + r) * (size_t)ldc + n] = acc[mi][ni][r] + bv;
        }
    }
}

// ---------------------------------------------------------------------------
// Elementwise kernels
// ---------------------------------------------------------------------------
__global__ void f32_to_bf16(const float* __restrict__ in, __bf16* __restrict__ out, int n)
{
    int i = blockIdx.x * blockDim.x + threadIdx.x;
    if (i < n) out[i] = (__bf16)in[i];
}

__global__ void embed_kernel(const int* __restrict__ input, const float* __restrict__ emb,
                             __bf16* __restrict__ x_bf)
{
    size_t i = (size_t)blockIdx.x * blockDim.x + threadIdx.x;
    if (i >= (size_t)kB * kS * kE) return;
    size_t bs = i / kE;
    int e = (int)(i % kE);
    int row = input[bs];
    x_bf[i] = (__bf16)emb[(size_t)row * kE + e];
}

__device__ inline float sigm(float x) { return 1.0f / (1.0f + expf(-x)); }

__global__ __launch_bounds__(256)
void lstm_cell(const float* __restrict__ gates, const float* __restrict__ c_in, int c_zero,
               float* __restrict__ c_out, float* __restrict__ h_f32,
               __bf16* __restrict__ h_bf, int ldh)
{
    size_t i = (size_t)blockIdx.x * blockDim.x + threadIdx.x;
    if (i >= (size_t)kB * kH) return;
    int b = (int)(i / kH), j = (int)(i % kH);
    const float* g = gates + (size_t)b * kG;
    float ig = sigm(g[j]);
    float fg = sigm(g[kH + j]);
    float gg = tanhf(g[2 * kH + j]);
    float og = sigm(g[3 * kH + j]);
    float c = c_zero ? 0.0f : c_in[i];
    c = fg * c + ig * gg;
    float h = og * tanhf(c);
    c_out[i] = c;
    if (h_f32) h_f32[i] = h;
    h_bf[(size_t)b * ldh + j] = (__bf16)h;
}

__global__ void dec_init(const float* __restrict__ h_enc, float* __restrict__ c_dec,
                         int* __restrict__ mask)
{
    size_t i = (size_t)blockIdx.x * blockDim.x + threadIdx.x;
    if (i < (size_t)kB * kH) c_dec[i] = h_enc[i];          // c0 = last encoder h
    if (i < (size_t)kB * kS) mask[i] = 0;
}

// One block per batch row: scores_s = vt . tanh(blend1[b,s,:] + blend2[b,:]),
// masked log-softmax over S, categorical sample, mask update.
__global__ __launch_bounds__(256)
void attn_kernel(const float* __restrict__ blend1, const float* __restrict__ blend2,
                 const float* __restrict__ vt, int* __restrict__ mask,
                 float* __restrict__ out_probs, float* __restrict__ out_tour, int step)
{
    __shared__ float red[256];
    __shared__ float sc[kS];
    const int b = blockIdx.x, tid = threadIdx.x;
    const float b2 = blend2[(size_t)b * kW + tid];
    const float vw = vt[tid];
    for (int s = 0; s < kS; ++s) {
        float v = tanhf(blend1[((size_t)b * kS + s) * kW + tid] + b2) * vw;
        red[tid] = v;
        __syncthreads();
        for (int off = 128; off > 0; off >>= 1) {
            if (tid < off) red[tid] += red[tid + off];
            __syncthreads();
        }
        if (tid == 0) sc[s] = red[0];
        __syncthreads();
    }
    if (tid == 0) {
        float mx = -1e30f;
        for (int s = 0; s < kS; ++s) {
            float x = mask[(size_t)b * kS + s] ? -100000.0f : sc[s];
            sc[s] = x;
            mx = fmaxf(mx, x);
        }
        float sum = 0.0f;
        for (int s = 0; s < kS; ++s) sum += expf(sc[s] - mx);
        float logZ = mx + logf(sum);
        for (int s = 0; s < kS; ++s)
            out_probs[((size_t)b * kL + step) * kS + s] = sc[s] - logZ;
        // counter-based hash RNG (stands in for threefry categorical)
        unsigned x = (unsigned)b * 1664525u ^ (unsigned)step * 1013904223u ^ 0x9E3779B9u;
        x ^= x >> 16; x *= 0x7feb352du; x ^= x >> 15; x *= 0x846ca68bu; x ^= x >> 16;
        float u = (float)(x >> 8) * (1.0f / 16777216.0f);
        float cum = 0.0f;
        int chosen = kS - 1;
        for (int s = 0; s < kS; ++s) {
            cum += expf(sc[s] - logZ);
            if (u <= cum) { chosen = s; break; }
        }
        mask[(size_t)b * kS + chosen] = 1;
        out_tour[(size_t)b * kL + step] = (float)chosen;
    }
}

// ---------------------------------------------------------------------------
// Host driver
// ---------------------------------------------------------------------------
extern "C" void kernel_launch(void* const* d_in, const int* in_sizes, int n_in,
                              void* d_out, int out_size, void* d_ws, size_t ws_size,
                              hipStream_t stream)
{
    const int*   input  = (const int*)  d_in[0];
    const float* emb    = (const float*)d_in[1];
    const float* encWih = (const float*)d_in[2];
    const float* encWhh = (const float*)d_in[3];
    const float* encBih = (const float*)d_in[4];
    const float* encBhh = (const float*)d_in[5];
    // d_in[6] (dec_Wih) unused: decoder input is identically zero in the reference
    const float* decWhh = (const float*)d_in[7];
    const float* decBih = (const float*)d_in[8];
    const float* decBhh = (const float*)d_in[9];
    const float* W1     = (const float*)d_in[10];
    const float* W2     = (const float*)d_in[11];
    const float* vt     = (const float*)d_in[12];

    float* out_probs = (float*)d_out;                       // [B, L, S] log-probs
    float* out_tour  = out_probs + (size_t)kB * kL * kS;    // [B, L] chosen idx

    char* ws = (char*)d_ws;
    size_t off = 0;
    auto alloc = [&](size_t bytes) -> void* {
        void* p = ws + off;
        off = (off + bytes + 255) & ~(size_t)255;
        return p;
    };

    __bf16* x_bf      = (__bf16*)alloc((size_t)kB * kS * kE * 2);
    __bf16* encWih_bf = (__bf16*)alloc((size_t)kG * kE * 2);
    __bf16* encWhh_bf = (__bf16*)alloc((size_t)kG * kH * 2);
    __bf16* decWhh_bf = (__bf16*)alloc((size_t)kG * kH * 2);
    __bf16* W1_bf     = (__bf16*)alloc((size_t)kW * kH * 2);
    __bf16* W2_bf     = (__bf16*)alloc((size_t)kW * kH * 2);
    __bf16* enc_bf    = (__bf16*)alloc((size_t)kB * kS * kH * 2); // encoder h states, bf16
    float*  h_enc     = (float*) alloc((size_t)kB * kH * 4);
    float*  c_buf     = (float*) alloc((size_t)kB * kH * 4);
    float*  gates     = (float*) alloc((size_t)kB * kG * 4);
    float*  blend1    = (float*) alloc((size_t)kB * kS * kW * 4);
    float*  blend2    = (float*) alloc((size_t)kB * kW * 4);
    __bf16* h_dec_bf  = (__bf16*)alloc((size_t)kB * kH * 2);
    int*    mask      = (int*)   alloc((size_t)kB * kS * 4);

    const dim3 blk(256);
    auto cdiv = [](size_t a, size_t b) { return (unsigned)((a + b - 1) / b); };

    // Weight conversions (tiny, L2-resident afterwards)
    f32_to_bf16<<<cdiv((size_t)kG * kE, 256), blk, 0, stream>>>(encWih, encWih_bf, kG * kE);
    f32_to_bf16<<<cdiv((size_t)kG * kH, 256), blk, 0, stream>>>(encWhh, encWhh_bf, kG * kH);
    f32_to_bf16<<<cdiv((size_t)kG * kH, 256), blk, 0, stream>>>(decWhh, decWhh_bf, kG * kH);
    f32_to_bf16<<<cdiv((size_t)kW * kH, 256), blk, 0, stream>>>(W1, W1_bf, kW * kH);
    f32_to_bf16<<<cdiv((size_t)kW * kH, 256), blk, 0, stream>>>(W2, W2_bf, kW * kH);

    embed_kernel<<<cdiv((size_t)kB * kS * kE, 256), blk, 0, stream>>>(input, emb, x_bf);

    // ---------------- Encoder: 50 dependent LSTM steps ----------------
    for (int t = 0; t < kS; ++t) {
        dim3 grid(kG / 128, kB / 128);
        gemm_bf16_wmma<<<grid, blk, 0, stream>>>(
            x_bf + (size_t)t * kE, kS * kE, encWih_bf, kE, kE,
            (t > 0) ? enc_bf + (size_t)(t - 1) * kH : nullptr, kS * kH,
            encWhh_bf, kH, (t > 0) ? kH : 0,
            encBih, encBhh, gates, kG);
        lstm_cell<<<cdiv((size_t)kB * kH, 256), blk, 0, stream>>>(
            gates, c_buf, (t == 0) ? 1 : 0, c_buf, h_enc,
            enc_bf + (size_t)t * kH, kS * kH);
    }

    // blend1 = enc_states @ W1^T : [B*S, W]
    {
        dim3 grid(kW / 128, (kB * kS) / 128);
        gemm_bf16_wmma<<<grid, blk, 0, stream>>>(
            enc_bf, kH, W1_bf, kH, kH,
            nullptr, 0, nullptr, 0, 0,
            nullptr, nullptr, blend1, kW);
    }

    // ---------------- Decoder: 50 attention/sampling steps ----------------
    dec_init<<<cdiv((size_t)kB * kH, 256), blk, 0, stream>>>(h_enc, c_buf, mask);
    for (int l = 0; l < kL; ++l) {
        dim3 grid(kG / 128, kB / 128);
        gemm_bf16_wmma<<<grid, blk, 0, stream>>>(
            (l > 0) ? h_dec_bf : nullptr, kH, decWhh_bf, kH, (l > 0) ? kH : 0,
            nullptr, 0, nullptr, 0, 0,
            decBih, decBhh, gates, kG);
        lstm_cell<<<cdiv((size_t)kB * kH, 256), blk, 0, stream>>>(
            gates, c_buf, 0, c_buf, nullptr, h_dec_bf, kH);
        dim3 grid2(kW / 128, kB / 128);
        gemm_bf16_wmma<<<grid2, blk, 0, stream>>>(
            h_dec_bf, kH, W2_bf, kH, kH,
            nullptr, 0, nullptr, 0, 0,
            nullptr, nullptr, blend2, kW);
        attn_kernel<<<kB, blk, 0, stream>>>(blend1, blend2, vt, mask,
                                            out_probs, out_tour, l);
    }
}
